// AnchorRefine_61856118997272
// MI455X (gfx1250) — compile-verified, matrix operand order checked by reading, not run on hardware
//
#include <hip/hip_runtime.h>
#include <hip/hip_bf16.h>
#include <math.h>

// ---------------- problem constants (match reference) ----------------
#define A_TOTAL 36864      // anchors (64*64*9)
#define N_IMG   32
#define M_GT    32
#define K_TOTAL 256
#define MAX_FG  128
#define IMG_SZ  1024.0f
#define POS_OV  0.7f
#define NEG_OV  0.3f

typedef __attribute__((ext_vector_type(16))) _Float16 v16h;
typedef __attribute__((ext_vector_type(8)))  float    v8f;

// ---------------- workspace layout (uint words) ----------------
#define W_PERGT   0                        // N*M monotonic max-keys of ov per gt
#define W_FGCNT   (W_PERGT + N_IMG*M_GT)   // N fg counts
#define W_BGCNT   (W_FGCNT + N_IMG)        // N bg counts
#define W_HIST    (W_BGCNT + N_IMG)        // N*256 radix histogram
#define W_SELPREF (W_HIST + N_IMG*256)     // radix prefix per image
#define W_SELK    (W_SELPREF + N_IMG)      // remaining-k per image
#define W_SELKA   (W_SELK + N_IMG)         // keep-all flag per image
#define W_FTHR    (W_SELKA + N_IMG)        // fg threshold bits
#define W_FEK     (W_FTHR + N_IMG)         // fg equal-keep quota
#define W_BTHR    (W_FEK + N_IMG)          // bg threshold bits
#define W_BEK     (W_BTHR + N_IMG)         // bg equal-keep quota
#define CTRL_WORDS (W_BEK + N_IMG)

// ---------------- CDNA5 async global->LDS copy (inline asm, ASYNCcnt) -------
// VDST carries the LDS byte offset; a generic pointer to a __shared__ object is
// (aperture_hi32 | lds_offset), so the low 32 bits are the LDS offset.
__device__ __forceinline__ void async_ld_b32(void* lds_ptr, const void* gptr) {
    unsigned lds_off = (unsigned)(size_t)lds_ptr;
    unsigned long long ga = (unsigned long long)(size_t)gptr;
    asm volatile("global_load_async_to_lds_b32 %0, %1, off"
                 :: "v"(lds_off), "v"(ga) : "memory");
}
__device__ __forceinline__ void wait_async0() {
    asm volatile("s_wait_asynccnt 0x0" ::: "memory");
}

// ---------------- helpers ----------------
// monotonic uint key for (possibly negative) floats: order(key) == order(float)
__device__ __forceinline__ unsigned ovkey(float f) {
    unsigned u = __float_as_uint(f);
    return (u & 0x80000000u) ? ~u : (u | 0x80000000u);
}

// bbox_transform for one anchor (must be bit-identical in phase1 & phase2)
__device__ __forceinline__ void make_pred(const float* __restrict__ an,
                                          const float* __restrict__ de,
                                          float* pb, bool* valid) {
    float a0 = an[0], a1 = an[1], a2 = an[2], a3 = an[3];
    float w  = a2 - a0 + 1.0f, h = a3 - a1 + 1.0f;
    float cx = a0 + 0.5f * w,  cy = a1 + 0.5f * h;
    float pcx = de[0] * w + cx;
    float pcy = de[1] * h + cy;
    float pw  = expf(de[2]) * w;
    float ph  = expf(de[3]) * h;
    pb[0] = pcx - 0.5f * pw;  pb[1] = pcy - 0.5f * ph;
    pb[2] = pcx + 0.5f * pw;  pb[3] = pcy + 0.5f * ph;
    *valid = (pb[0] >= 0.0f) && (pb[1] >= 0.0f) && (pb[2] < IMG_SZ) && (pb[3] < IMG_SZ);
}

__device__ __forceinline__ float iou_ov(const float* pb, const float* g, bool valid) {
    if (!valid) return -1.0f;
    float area_a = (pb[2]-pb[0]+1.0f) * (pb[3]-pb[1]+1.0f);
    float area_g = (g[2]-g[0]+1.0f) * (g[3]-g[1]+1.0f);
    float iw = fminf(pb[2], g[2]) - fmaxf(pb[0], g[0]) + 1.0f;
    float ih = fminf(pb[3], g[3]) - fmaxf(pb[1], g[1]) + 1.0f;
    iw = fmaxf(iw, 0.0f); ih = fmaxf(ih, 0.0f);
    float inter = iw * ih;
    return inter / (area_a + area_g - inter);
}

// Wave-level popcount via the matrix unit: a sum-reduction is layout-invariant,
// so each lane drops its flag into one A-matrix cell, B = ones, and the f32
// accumulator row-sums give the wave total.  Emits v_wmma_f32_16x16x32_f16.
// Requires all 32 lanes active (EXEC == ~0).
__device__ __forceinline__ float wave_count32(bool flag) {
    v16h a;
#pragma unroll
    for (int i = 0; i < 16; ++i) a[i] = (_Float16)0.0f;
    a[0] = flag ? (_Float16)1.0f : (_Float16)0.0f;
    v16h b;
#pragma unroll
    for (int i = 0; i < 16; ++i) b[i] = (_Float16)1.0f;
    v8f c = {};
    c = __builtin_amdgcn_wmma_f32_16x16x32_f16(false, a, false, b, (short)0, c, false, false);
    // C/D layout: lane<16 holds rows 0..7 (vgpr v -> M=v), lane>=16 rows 8..15.
    float s = 0.0f;
#pragma unroll
    for (int i = 0; i < 8; ++i) s += c[i];
    s += __shfl_xor(s, 16, 32);     // rows 0..7 + rows 8..15 -> full total
    return s;
}

// ---------------- kernels ----------------
__global__ void k_init(unsigned* __restrict__ ws) {
    int i = blockIdx.x * 256 + threadIdx.x;
    if (i < CTRL_WORDS) ws[i] = 0u;
}

// Pass 1: per-gt column max of ov (monotone keys, LDS-combined atomics)
__global__ void k_phase1(const float* __restrict__ anchors,
                         const float* __restrict__ gt,
                         const float* __restrict__ deltas,
                         unsigned* __restrict__ ws) {
    __shared__ float    sgt[M_GT * 4];
    __shared__ unsigned smax[M_GT];
    int img = blockIdx.y;
    int a   = blockIdx.x * 256 + threadIdx.x;
    // stage gt boxes via the async global->LDS path (ASYNCcnt)
    if (threadIdx.x < M_GT * 4)
        async_ld_b32(&sgt[threadIdx.x], &gt[img * M_GT * 4 + threadIdx.x]);
    if (threadIdx.x < M_GT) smax[threadIdx.x] = 0u;
    wait_async0();
    __syncthreads();

    // gfx1250 global prefetch of the next block's deltas (speculative, in-range only)
    if (blockIdx.x + 1 < gridDim.x)
        __builtin_prefetch(&deltas[((size_t)img * A_TOTAL + (size_t)(a + 256)) * 4], 0, 1);

    float pb[4]; bool valid;
    make_pred(&anchors[(size_t)a * 4], &deltas[((size_t)img * A_TOTAL + a) * 4], pb, &valid);
#pragma unroll 4
    for (int m = 0; m < M_GT; ++m) {
        float ov = iou_ov(pb, &sgt[m * 4], valid);
        atomicMax(&smax[m], ovkey(ov));
    }
    __syncthreads();
    if (threadIdx.x < M_GT)
        atomicMax(&ws[W_PERGT + img * M_GT + threadIdx.x], smax[threadIdx.x]);
}

// Pass 2: labels + gt argmax + fg/bg counts (WMMA reduction)
__global__ void k_phase2(const float* __restrict__ anchors,
                         const float* __restrict__ gt,
                         const float* __restrict__ deltas,
                         unsigned* __restrict__ ws,
                         signed char* __restrict__ labels,
                         unsigned char* __restrict__ gtid) {
    __shared__ float    sgt[M_GT * 4];
    __shared__ unsigned spg[M_GT];
    int img = blockIdx.y;
    int a   = blockIdx.x * 256 + threadIdx.x;
    if (threadIdx.x < M_GT * 4)
        async_ld_b32(&sgt[threadIdx.x], &gt[img * M_GT * 4 + threadIdx.x]);
    if (threadIdx.x < M_GT)
        async_ld_b32(&spg[threadIdx.x], &ws[W_PERGT + img * M_GT + threadIdx.x]);
    wait_async0();
    __syncthreads();

    float pb[4]; bool valid;
    make_pred(&anchors[(size_t)a * 4], &deltas[((size_t)img * A_TOTAL + a) * 4], pb, &valid);

    float amax = -2.0f; int bm = 0; bool abox = false;
#pragma unroll 4
    for (int m = 0; m < M_GT; ++m) {
        float ov = iou_ov(pb, &sgt[m * 4], valid);
        if (ov > amax) { amax = ov; bm = m; }           // first-max like jnp.argmax
        if (valid && ovkey(ov) == spg[m]) abox = true;  // ov == per_gt_max
    }
    bool bbox = amax >= POS_OV;
    bool neg  = valid && (amax < NEG_OV);
    int lab = (abox || bbox) ? 1 : (neg ? 0 : -1);

    size_t off = (size_t)img * A_TOTAL + a;
    labels[off] = (signed char)lab;
    gtid[off]   = (unsigned char)bm;

    // matrix-unit popcount (all lanes active here)
    float cf = wave_count32(lab == 1);
    float cb = wave_count32(lab == 0);
    if ((threadIdx.x & 31) == 0) {
        atomicAdd(&ws[W_FGCNT + img], (unsigned)(cf + 0.5f));
        atomicAdd(&ws[W_BGCNT + img], (unsigned)(cb + 0.5f));
    }
}

// Radix-select init: decide keep-all vs select; K_fg=128, K_bg=256-min(F,128)
__global__ void k_sel_init(unsigned* __restrict__ ws, int isFg) {
    int img = threadIdx.x;
    if (img >= N_IMG) return;
    unsigned C, K;
    if (isFg) { C = ws[W_FGCNT + img]; K = MAX_FG; }
    else {
        unsigned f = ws[W_FGCNT + img];
        unsigned fk = f < MAX_FG ? f : MAX_FG;
        C = ws[W_BGCNT + img];
        K = K_TOTAL - fk;
    }
    ws[W_SELPREF + img] = 0u;
    if (C <= K || K == 0u) {
        ws[W_SELKA + img] = 1u;
        ws[W_SELK  + img] = 0u;
        unsigned thr = 0u, ek = 0x7FFFFFFFu;     // keep-all encoding
        if (K == 0u) { thr = 0xFFFFFFFFu; ek = 0u; }  // keep-none encoding
        if (isFg) { ws[W_FTHR + img] = thr; ws[W_FEK + img] = ek; }
        else      { ws[W_BTHR + img] = thr; ws[W_BEK + img] = ek; }
    } else {
        ws[W_SELKA + img] = 0u;
        ws[W_SELK  + img] = K;
    }
}

// Radix histogram of one byte of the (non-negative) score bits
__global__ void k_hist(const float* __restrict__ rs,
                       const signed char* __restrict__ labels,
                       unsigned* __restrict__ ws, int target, int shift) {
    int img = blockIdx.y;
    if (ws[W_SELKA + img]) return;
    int a = blockIdx.x * 256 + threadIdx.x;
    size_t off = (size_t)img * A_TOTAL + a;
    if ((int)labels[off] != target) return;
    unsigned key = __float_as_uint(rs[off]);     // r in [0,1): bits are order-preserving
    unsigned himask = (shift >= 24) ? 0u : (0xFFFFFFFFu << (shift + 8));
    if ((key & himask) != ws[W_SELPREF + img]) return;
    atomicAdd(&ws[W_HIST + img * 256 + ((key >> shift) & 255u)], 1u);
}

// Pick the bucket holding the k-th largest, update prefix/remaining-k, zero hist
__global__ void k_select(unsigned* __restrict__ ws, int isFg, int shift) {
    __shared__ unsigned cnt[256];
    int img = blockIdx.x;
    cnt[threadIdx.x] = ws[W_HIST + img * 256 + threadIdx.x];
    __syncthreads();
    if (threadIdx.x == 0 && !ws[W_SELKA + img]) {
        unsigned kRem = ws[W_SELK + img];
        unsigned pref = ws[W_SELPREF + img];
        unsigned acc = 0; int chosen = 0;
        for (int b = 255; b >= 0; --b) {
            unsigned c = cnt[b];
            if (acc + c >= kRem) { chosen = b; kRem -= acc; break; }
            acc += c;
        }
        pref |= ((unsigned)chosen) << shift;
        ws[W_SELPREF + img] = pref;
        ws[W_SELK + img]    = kRem;           // rank within chosen bucket
        if (shift == 0) {                     // final: thr = k-th largest key
            if (isFg) { ws[W_FTHR + img] = pref; ws[W_FEK + img] = kRem; }
            else      { ws[W_BTHR + img] = pref; ws[W_BEK + img] = kRem; }
        }
    }
    __syncthreads();
    ws[W_HIST + img * 256 + threadIdx.x] = 0u;   // ready for next pass
}

__device__ __forceinline__ void write_slot(float* oIdx, float* oFg, float* oCf,
                                           int img, unsigned pos, int a, bool fg, int g,
                                           const float* __restrict__ anchors,
                                           const float* __restrict__ gtb) {
    int s = img * K_TOTAL + (int)pos;
    oIdx[s] = (float)a;
    oFg[s]  = fg ? 1.0f : 0.0f;
    const float* an = &anchors[(size_t)a * 4];
    const float* gb = &gtb[((size_t)img * M_GT + g) * 4];
    float aw = an[2]-an[0]+1.0f, ah = an[3]-an[1]+1.0f;
    float acx = an[0]+0.5f*aw,   acy = an[1]+0.5f*ah;
    float gw = gb[2]-gb[0]+1.0f, gh = gb[3]-gb[1]+1.0f;
    float gcx = gb[0]+0.5f*gw,   gcy = gb[1]+0.5f*gh;
    oCf[(size_t)s*4+0] = (gcx-acx)/aw;
    oCf[(size_t)s*4+1] = (gcy-acy)/ah;
    oCf[(size_t)s*4+2] = logf(gw/aw);
    oCf[(size_t)s*4+3] = logf(gh/ah);
}

// One wave32 per image: index-ordered keep decisions (stable tie-break via
// ballot prefix), stream compaction, and target-coefficient computation.
__global__ void k_compact(const float* __restrict__ anchors,
                          const float* __restrict__ gt,
                          const float* __restrict__ rs,
                          const signed char* __restrict__ labels,
                          const unsigned char* __restrict__ gtid,
                          const unsigned* __restrict__ ws,
                          float* __restrict__ out) {
    int img = blockIdx.x, lane = threadIdx.x;         // 32 threads = 1 wave
    unsigned fthr = ws[W_FTHR + img], fek = ws[W_FEK + img];
    unsigned bthr = ws[W_BTHR + img], bek = ws[W_BEK + img];
    float* oIdx = out;
    float* oFg  = out + (size_t)N_IMG * K_TOTAL;
    float* oCf  = out + (size_t)2 * N_IMG * K_TOTAL;
    unsigned keptBase = 0, eqF = 0, eqB = 0;
    unsigned lmask = (1u << lane) - 1u;
    float firstFg = 0.0f; int firstGt = 0;

    for (int i = 0; i < A_TOTAL; i += 32) {
        int a = i + lane;
        size_t off = (size_t)img * A_TOTAL + a;
        int lab = (int)labels[off];
        unsigned u = __float_as_uint(rs[off]);
        int g = (int)gtid[off];
        bool isf = (lab == 1), isb = (lab == 0);

        bool kfe = isf && (u == fthr);
        unsigned balF = (unsigned)__ballot(kfe);
        bool keepf = (isf && u > fthr) || (kfe && (eqF + __popc(balF & lmask)) < fek);
        eqF += __popc(balF);

        bool kbe = isb && (u == bthr);
        unsigned balB = (unsigned)__ballot(kbe);
        bool keepb = (isb && u > bthr) || (kbe && (eqB + __popc(balB & lmask)) < bek);
        eqB += __popc(balB);

        bool kept = keepf || keepb;
        unsigned kb = (unsigned)__ballot(kept);
        unsigned pos = keptBase + __popc(kb & lmask);
        if (kept && pos < K_TOTAL)
            write_slot(oIdx, oFg, oCf, img, pos, a, keepf, g, anchors, gt);
        keptBase += __popc(kb);

        if (i == 0) {   // reference's zero-scatter default slot uses anchor 0
            firstFg = __shfl(keepf ? 1.0f : 0.0f, 0, 32);
            firstGt = __shfl(g, 0, 32);
        }
    }
    for (unsigned p = keptBase + lane; p < K_TOTAL; p += 32)
        write_slot(oIdx, oFg, oCf, img, p, 0, firstFg > 0.5f, firstGt, anchors, gt);
}

// ---------------- launch ----------------
extern "C" void kernel_launch(void* const* d_in, const int* in_sizes, int n_in,
                              void* d_out, int out_size, void* d_ws, size_t ws_size,
                              hipStream_t stream) {
    const float* anchors = (const float*)d_in[0];   // [A,4]
    const float* gt      = (const float*)d_in[1];   // [N,M,4]
    const float* deltas  = (const float*)d_in[2];   // [N,A,4]
    const float* rs      = (const float*)d_in[3];   // [N,A]
    float* out = (float*)d_out;                     // idx | fg | coeff (floats)

    unsigned* ws = (unsigned*)d_ws;
    signed char*   labels = (signed char*)((char*)d_ws + (size_t)CTRL_WORDS * 4);
    unsigned char* gtid   = (unsigned char*)((char*)d_ws + (size_t)CTRL_WORDS * 4
                                             + (size_t)N_IMG * A_TOTAL);

    dim3 gA(A_TOTAL / 256, N_IMG);

    k_init<<<(CTRL_WORDS + 255) / 256, 256, 0, stream>>>(ws);
    k_phase1<<<gA, 256, 0, stream>>>(anchors, gt, deltas, ws);
    k_phase2<<<gA, 256, 0, stream>>>(anchors, gt, deltas, ws, labels, gtid);

    k_sel_init<<<1, 32, 0, stream>>>(ws, 1);                       // fg: K = 128
    for (int shift = 24; shift >= 0; shift -= 8) {
        k_hist<<<gA, 256, 0, stream>>>(rs, labels, ws, 1, shift);
        k_select<<<N_IMG, 256, 0, stream>>>(ws, 1, shift);
    }
    k_sel_init<<<1, 32, 0, stream>>>(ws, 0);                       // bg: K = 256-min(F,128)
    for (int shift = 24; shift >= 0; shift -= 8) {
        k_hist<<<gA, 256, 0, stream>>>(rs, labels, ws, 0, shift);
        k_select<<<N_IMG, 256, 0, stream>>>(ws, 0, shift);
    }
    k_compact<<<N_IMG, 32, 0, stream>>>(anchors, gt, rs, labels, gtid, ws, out);
}
